// Model_65524021067891
// MI455X (gfx1250) — matrix operand order repaired and checked by reading, not compiled, not run on hardware
//
#include <hip/hip_runtime.h>
#include <hip/hip_bf16.h>

#define VOCAB   32000
#define EMBED   512
#define HIDDEN  1024
#define G3      (3 * HIDDEN)
#define BATCH   256
#define SEQT    512
#define TILE_B  16
#define NWAVES  16
#define BLOCK   (NWAVES * 32)   // 512 threads = 16 wave32 waves (4/SIMD, ~256 VGPRs/wave)
#define SUBT    4               // sub-tiles of 16 hidden cols per wave (64 cols/wave)

typedef __attribute__((ext_vector_type(16))) _Float16 v16h;
typedef __attribute__((ext_vector_type(8)))  _Float16 v8h;
typedef __attribute__((ext_vector_type(8)))  float    v8f;

union HFrag { v16h v; v8h h[2]; };

__device__ __forceinline__ float sigmoidf_(float x) {
    return 1.0f / (1.0f + __expf(-x));
}

// ---- prep: f32 -> f16 weight conversion (runs every launch; deterministic) ----
__global__ void cvt_f32_f16(const float* __restrict__ src,
                            _Float16* __restrict__ dst, int n) {
    int i = blockIdx.x * blockDim.x + threadIdx.x;
    int stride = gridDim.x * blockDim.x;
    for (; i < n; i += stride) dst[i] = (_Float16)src[i];
}

// Gather the 16 token embeddings for step t into LDS as f16.
// 512 threads: thread handles row tid>>5, 16 consecutive elements.
__device__ __forceinline__ void gather_x(const int* __restrict__ x,
                                         const float* __restrict__ emb,
                                         _Float16* __restrict__ xH,
                                         int b0, int t, int tid) {
    const int r    = tid >> 5;
    const int lane = tid & 31;
    const int tok  = x[(b0 + r) * SEQT + t];
    const float* src = emb + (size_t)tok * EMBED + lane * 16;
    _Float16*    dst = xH + r * EMBED + lane * 16;
#pragma unroll
    for (int q = 0; q < 16; q += 4) {
        float4 f = *(const float4*)(src + q);
        dst[q + 0] = (_Float16)f.x;
        dst[q + 1] = (_Float16)f.y;
        dst[q + 2] = (_Float16)f.z;
        dst[q + 3] = (_Float16)f.w;
    }
}

// Accumulate a 16x16 output tile over K for the 3 gate column-slices.
// A: LDS f16 [16 x astride], rows = batch tile rows. B: f16 weights, row g = K-run.
__device__ __forceinline__ void mm_accum(const _Float16* __restrict__ Abase, int astride,
                                         const _Float16* __restrict__ Br,
                                         const _Float16* __restrict__ Bz,
                                         const _Float16* __restrict__ Bn,
                                         int K, int arow, int ahalf, int bhalf,
                                         v8f& aR, v8f& aZ, v8f& aN) {
    for (int kt = 0; kt < K; kt += 32) {
        HFrag a, fr, fz, fn;
        // A frag: lanes 0-15 hold K=kt+0..15 (row = lane),
        //         lanes 16-31 hold K=kt+16..31 (2*ahalf = 16)
        const _Float16* ap = Abase + arow * astride + kt + 2 * ahalf;
        a.h[0] = *(const v8h*)(ap);
        a.h[1] = *(const v8h*)(ap + 8);
        // B frag: lane holds 16 consecutive K for its column; lanes>=16 take K+16
        const int klo = kt + bhalf;
        fr.h[0] = *(const v8h*)(Br + klo);  fr.h[1] = *(const v8h*)(Br + klo + 8);
        fz.h[0] = *(const v8h*)(Bz + klo);  fz.h[1] = *(const v8h*)(Bz + klo + 8);
        fn.h[0] = *(const v8h*)(Bn + klo);  fn.h[1] = *(const v8h*)(Bn + klo + 8);
        aR = __builtin_amdgcn_wmma_f32_16x16x32_f16(false, a.v, false, fr.v, (short)0, aR, false, false);
        aZ = __builtin_amdgcn_wmma_f32_16x16x32_f16(false, a.v, false, fz.v, (short)0, aZ, false, false);
        aN = __builtin_amdgcn_wmma_f32_16x16x32_f16(false, a.v, false, fn.v, (short)0, aN, false, false);
    }
}

__global__ void __launch_bounds__(BLOCK, 1)
gru_scan(const int* __restrict__ x, const float* __restrict__ h0,
         const int* __restrict__ seq_len, const float* __restrict__ emb,
         const _Float16* __restrict__ wih, const _Float16* __restrict__ whh,
         const float* __restrict__ b_ih, const float* __restrict__ b_hh,
         const float* __restrict__ wlin_g, const float* __restrict__ blin,
         float* __restrict__ out) {
    extern __shared__ char smem[];
    float*    hF  = (float*)smem;                         // 64 KB: f32 master h [16][1024]
    _Float16* hHb[2];
    hHb[0] = (_Float16*)(smem + 64 * 1024);               // 32 KB: f16 h, buffer 0
    hHb[1] = (_Float16*)(smem + 96 * 1024);               // 32 KB: f16 h, buffer 1
    _Float16* xH  = (_Float16*)(smem + 128 * 1024);       // 16 KB: f16 x_t [16][512]
    float*    wl  = (float*)(smem + 144 * 1024);          //  4 KB: W_lin [1024]

    const int tid  = threadIdx.x;
    const int wave = tid >> 5;              // 0..15
    const int lane = tid & 31;
    const int b0   = blockIdx.x * TILE_B;

    // init h (f32 + f16 copies), W_lin, and x embeddings for t=0
    for (int i = tid; i < TILE_B * HIDDEN; i += BLOCK) {
        float v = h0[(size_t)b0 * HIDDEN + i];
        hF[i]     = v;
        hHb[0][i] = (_Float16)v;
    }
    for (int i = tid; i < HIDDEN; i += BLOCK) wl[i] = wlin_g[i];
    gather_x(x, emb, xH, b0, 0, tid);

    const int   sl  = seq_len[b0 + wave];   // wave w owns batch row w for the head
    const float bl0 = blin[0];

    const int arow  = lane & 15;            // A-frag M row
    const int ahalf = (lane >> 4) * 8;      // A-frag K sub-offset
    const int ncol  = lane & 15;            // B / C / D column within 16-tile
    const int bhalf = (lane >> 4) * 16;     // B-frag K offset
    const int mbase = (lane >> 4) * 8;      // C/D M base (VGPR p -> M = p + mbase)

    // hoist loop-invariant gate biases (per wave, per sub-tile) out of the scan
    float biR[SUBT], biZ[SUBT], biN[SUBT], bhN[SUBT];
#pragma unroll
    for (int s = 0; s < SUBT; ++s) {
        const int c0 = wave * (16 * SUBT) + s * 16;
        const int gr = c0 + ncol;
        const int gz = HIDDEN + c0 + ncol;
        const int gn = 2 * HIDDEN + c0 + ncol;
        biR[s] = b_ih[gr] + b_hh[gr];
        biZ[s] = b_ih[gz] + b_hh[gz];
        biN[s] = b_ih[gn];
        bhN[s] = b_hh[gn];
    }

    __syncthreads();

    for (int t = 0; t < SEQT; ++t) {
        const _Float16* hHc = hHb[t & 1];         // current f16 h (read by WMMA)
        _Float16*       hHn = hHb[(t & 1) ^ 1];   // next f16 h (written below)

        // --- per wave: SUBT sub-tiles of 16 hidden columns each ---
        for (int s = 0; s < SUBT; ++s) {
            const int c0 = wave * (16 * SUBT) + s * 16;
            const int gr = c0 + ncol;               // r-gate column
            const int gz = HIDDEN + c0 + ncol;      // z-gate column
            const int gn = 2 * HIDDEN + c0 + ncol;  // n-gate column
            v8f aR = {}, aZ = {}, aI = {}, aH = {};

            // gi = x_t @ W_ih^T   (K = 512); n-gate input part kept separate (aI)
            mm_accum(xH, EMBED,
                     wih + (size_t)gr * EMBED, wih + (size_t)gz * EMBED,
                     wih + (size_t)gn * EMBED,
                     EMBED, arow, ahalf, bhalf, aR, aZ, aI);
            // gh = h @ W_hh^T     (K = 1024); n-gate hidden part kept separate (aH)
            mm_accum(hHc, HIDDEN,
                     whh + (size_t)gr * HIDDEN, whh + (size_t)gz * HIDDEN,
                     whh + (size_t)gn * HIDDEN,
                     HIDDEN, arow, ahalf, bhalf, aR, aZ, aH);

            const int col = c0 + ncol;
            // C/D layout: VGPR p holds (M = p + mbase, N = ncol)
#pragma unroll
            for (int p = 0; p < 8; ++p) {
                const int m  = mbase + p;
                float r    = sigmoidf_(aR[p] + biR[s]);
                float zg   = sigmoidf_(aZ[p] + biZ[s]);
                float nn   = tanhf(aI[p] + biN[s] + r * (aH[p] + bhN[s]));
                float hold = hF[m * HIDDEN + col];          // owner-exclusive RMW
                float hnew = (1.0f - zg) * nn + zg * hold;
                hF[m * HIDDEN + col]  = hnew;
                hHn[m * HIDDEN + col] = (_Float16)hnew;     // next f16 buffer
            }
        }
        __syncthreads();   // all WMMA reads of hHc/xH and all hF/hHn writes done

        // prefetch next step's token embeddings (xH free to overwrite now)
        if (t + 1 < SEQT) gather_x(x, emb, xH, b0, t + 1, tid);

        // head: wave w reduces row w when this is its last valid step
        if (sl - 1 == t) {
            float acc = 0.0f;
            for (int j = lane; j < HIDDEN; j += 32)
                acc += hF[wave * HIDDEN + j] * wl[j];
#pragma unroll
            for (int off = 16; off > 0; off >>= 1)
                acc += __shfl_xor(acc, off, 32);
            if (lane == 0)
                out[b0 + wave] = sigmoidf_(acc + bl0);
        }
        __syncthreads();   // xH stable & head reads done before next step mutates hF
    }

    // h_final (f32) -> d_out[256 + b*H + j]
    for (int i = tid; i < TILE_B * HIDDEN; i += BLOCK)
        out[BATCH + (size_t)b0 * HIDDEN + i] = hF[i];
}

extern "C" void kernel_launch(void* const* d_in, const int* in_sizes, int n_in,
                              void* d_out, int out_size, void* d_ws, size_t ws_size,
                              hipStream_t stream) {
    const int*   x     = (const int*)d_in[0];
    const float* h0    = (const float*)d_in[1];
    const int*   slen  = (const int*)d_in[2];
    const float* emb   = (const float*)d_in[3];
    const float* W_ih  = (const float*)d_in[4];
    const float* W_hh  = (const float*)d_in[5];
    const float* b_ih  = (const float*)d_in[6];
    const float* b_hh  = (const float*)d_in[7];
    const float* W_lin = (const float*)d_in[8];
    const float* b_lin = (const float*)d_in[9];
    float* out = (float*)d_out;

    // ws layout: f16 W_ih (3 MB) | f16 W_hh (6 MB)
    _Float16* wih16 = (_Float16*)d_ws;
    _Float16* whh16 = wih16 + (size_t)G3 * EMBED;

    cvt_f32_f16<<<1024, 256, 0, stream>>>(W_ih, wih16, G3 * EMBED);
    cvt_f32_f16<<<1024, 256, 0, stream>>>(W_hh, whh16, G3 * HIDDEN);

    // LDS: 64K hF + 2*32K hH + 16K xH + 4K W_lin = 148 KB (<320 KB/WGP)
    const size_t smem_bytes = 148 * 1024;
    gru_scan<<<BATCH / TILE_B, BLOCK, smem_bytes, stream>>>(
        x, h0, slen, emb, wih16, whh16, b_ih, b_hh, W_lin, b_lin, out);

    (void)in_sizes; (void)n_in; (void)out_size; (void)ws_size;
}
